// FaceNormalLoss_12060268167280
// MI455X (gfx1250) — compile-verified, matrix operand order checked by reading, not run on hardware
//
#include <hip/hip_runtime.h>
#include <hip/hip_bf16.h>

typedef __attribute__((ext_vector_type(16))) _Float16     v16h;
typedef __attribute__((ext_vector_type(8)))  float        v8f;
typedef __attribute__((ext_vector_type(8)))  unsigned int v8u;

#define NUM_BATCHES 32
#define WAVES_PER_BLOCK 8
#define BLOCK_SIZE 256
#define MAIN_BLOCKS 1024
// stage layout per wave: values[16 ch][32 faces] halfs, then segs[32] u16
#define STAGE_HALFS (16 * 32 + 32)
#define SEG_OFF (16 * 32)
// ws layout: mesh1 [32 batches][16 ch] floats at 0, mesh2 at 512
#define WS_M2 512
#define WS_TOTAL 1024

__device__ __forceinline__ unsigned short f2h_bits(float f) {
  _Float16 h = (_Float16)f;
  return __builtin_bit_cast(unsigned short, h);
}

// Per-face unit normal + batch id (int32 inputs per harness convention).
__device__ __forceinline__ void face_normal(const float* __restrict__ x,
                                            const int* __restrict__ b,
                                            const int* __restrict__ f,
                                            int F, int face,
                                            float& nx, float& ny, float& nz, int& seg) {
  int i0 = f[face];
  int i1 = f[F + face];
  int i2 = f[2 * F + face];
  float ax = x[3 * i0], ay = x[3 * i0 + 1], az = x[3 * i0 + 2];
  float bx = x[3 * i1], by = x[3 * i1 + 1], bz = x[3 * i1 + 2];
  float cx = x[3 * i2], cy = x[3 * i2 + 1], cz = x[3 * i2 + 2];
  float e1x = bx - ax, e1y = by - ay, e1z = bz - az;
  float e2x = cx - ax, e2y = cy - ay, e2z = cz - az;
  float vx = e1y * e2z - e1z * e2y;
  float vy = e1z * e2x - e1x * e2z;
  float vz = e1x * e2y - e1y * e2x;
  float len = sqrtf(vx * vx + vy * vy + vz * vz);
  float inv = 1.0f / (len + 1e-8f);
  nx = vx * inv; ny = vy * inv; nz = vz * inv;
  seg = b[i0];
}

// One-hot segment reduction of 32 staged faces via two f16 WMMAs (batches 0-15, 16-31).
__device__ __forceinline__ void wmma_segment_accum(const unsigned short* __restrict__ segS,
                                                   const unsigned short* __restrict__ valS,
                                                   int lane, v8f& acc_lo, v8f& acc_hi) {
  const int grp = (lane >> 4) & 1;
  const int m = lane & 15;          // batch row for A, channel column for B
  const unsigned int m16 = (unsigned)m + 16u;
  v8u aLo, aHi, bV;
#pragma unroll
  for (int v = 0; v < 8; ++v) {
    int k0 = 2 * (v & 3) + ((v >= 4) ? 16 : 0) + (grp ? 8 : 0);  // ISA 16-bit A/B K-pair map
    unsigned int sp = *(const unsigned int*)&segS[k0];           // seg[k0] | seg[k0+1]<<16
    unsigned int s0 = sp & 0xffffu, s1 = sp >> 16;
    aLo[v] = ((s0 == (unsigned)m) ? 0x3C00u : 0u) | ((s1 == (unsigned)m) ? 0x3C000000u : 0u);
    aHi[v] = ((s0 == m16) ? 0x3C00u : 0u) | ((s1 == m16) ? 0x3C000000u : 0u);
    bV[v]  = *(const unsigned int*)&valS[m * 32 + k0];           // column m, faces k0,k0+1
  }
  v16h A_lo = __builtin_bit_cast(v16h, aLo);
  v16h A_hi = __builtin_bit_cast(v16h, aHi);
  v16h B    = __builtin_bit_cast(v16h, bV);
  acc_lo = __builtin_amdgcn_wmma_f32_16x16x32_f16(false, A_lo, false, B, (short)0, acc_lo, false, false);
  acc_hi = __builtin_amdgcn_wmma_f32_16x16x32_f16(false, A_hi, false, B, (short)0, acc_hi, false, false);
}

__global__ void fn_init(float* __restrict__ ws) {
  int i = blockIdx.x * blockDim.x + threadIdx.x;
  if (i < WS_TOTAL) ws[i] = 0.0f;
}

__global__ __launch_bounds__(BLOCK_SIZE) void fn_main(
    const float* __restrict__ x1, const float* __restrict__ x2,
    const int* __restrict__ b1, const int* __restrict__ b2,
    const int* __restrict__ f1, const int* __restrict__ f2,
    float* __restrict__ ws, int F) {
  __shared__ __align__(16) unsigned short stage[WAVES_PER_BLOCK][STAGE_HALFS];

  const int tid = threadIdx.x;
  const int wave = tid >> 5;
  const int lane = tid & 31;
  unsigned short* valS = &stage[wave][0];
  unsigned short* segS = &stage[wave][SEG_OFF];

  const int nChunks = (F + 31) >> 5;
  const int gwave = blockIdx.x * WAVES_PER_BLOCK + wave;
  const int totWaves = gridDim.x * WAVES_PER_BLOCK;

  v8f acc1lo = {}, acc1hi = {}, acc2lo = {}, acc2hi = {};

  for (int c = gwave; c < nChunks; c += totWaves) {
    int face = (c << 5) + lane;
    bool valid = face < F;
    int fc = valid ? face : (F - 1);
    float vflag = valid ? 1.0f : 0.0f;
    float nx, ny, nz; int seg;

    // ---- mesh 1: channels {nx, ny, nz, 1} ----
    face_normal(x1, b1, f1, F, fc, nx, ny, nz, seg);
    valS[0 * 32 + lane] = f2h_bits(nx);
    valS[1 * 32 + lane] = f2h_bits(ny);
    valS[2 * 32 + lane] = f2h_bits(nz);
    valS[3 * 32 + lane] = f2h_bits(vflag);
    segS[lane] = (unsigned short)seg;
    asm volatile("s_wait_dscnt 0" ::: "memory");
    wmma_segment_accum(segS, valS, lane, acc1lo, acc1hi);
    asm volatile("" ::: "memory");

    // ---- mesh 2: channels {nx, ny, nz, nx^2, ny^2, nz^2, 1} ----
    face_normal(x2, b2, f2, F, fc, nx, ny, nz, seg);
    valS[0 * 32 + lane] = f2h_bits(nx);
    valS[1 * 32 + lane] = f2h_bits(ny);
    valS[2 * 32 + lane] = f2h_bits(nz);
    valS[3 * 32 + lane] = f2h_bits(nx * nx);
    valS[4 * 32 + lane] = f2h_bits(ny * ny);
    valS[5 * 32 + lane] = f2h_bits(nz * nz);
    valS[6 * 32 + lane] = f2h_bits(vflag);
    segS[lane] = (unsigned short)seg;
    asm volatile("s_wait_dscnt 0" ::: "memory");
    wmma_segment_accum(segS, valS, lane, acc2lo, acc2hi);
    asm volatile("" ::: "memory");
  }

  // Flush per-wave accumulators. D layout: VGPR r, lanes 0-15 -> (M=r, N=lane),
  // lanes 16-31 -> (M=r+8, N=lane-16).
  const int ch = lane & 15;
  const int rowAdd = (lane >> 4) ? 8 : 0;
#pragma unroll
  for (int r = 0; r < 8; ++r) {
    int batch = r + rowAdd;
    if (ch < 4) {
      atomicAdd(&ws[batch * 16 + ch], acc1lo[r]);
      atomicAdd(&ws[(batch + 16) * 16 + ch], acc1hi[r]);
    }
    if (ch < 7) {
      atomicAdd(&ws[WS_M2 + batch * 16 + ch], acc2lo[r]);
      atomicAdd(&ws[WS_M2 + (batch + 16) * 16 + ch], acc2hi[r]);
    }
  }
}

__global__ void fn_final(const float* __restrict__ ws, float* __restrict__ out) {
  int bat = threadIdx.x;  // 0..31, one wave
  const float* w1 = ws + bat * 16;
  const float* w2 = ws + WS_M2 + bat * 16;
  float cnt1 = w1[3];
  float cnt2 = w2[6];
  float m1x = w1[0] / cnt1, m1y = w1[1] / cnt1, m1z = w1[2] / cnt1;
  float m2x = w2[0] / cnt2, m2y = w2[1] / cnt2, m2z = w2[2] / cnt2;
  float denom = cnt2 - 1.0f;
  float vx = fmaxf(w2[3] - cnt2 * m2x * m2x, 0.0f) / denom;
  float vy = fmaxf(w2[4] - cnt2 * m2y * m2y, 0.0f) / denom;
  float vz = fmaxf(w2[5] - cnt2 * m2z * m2z, 0.0f) / denom;
  float consistency = sqrtf(vx + vy + vz);
  float dx = m1x - m2x, dy = m1y - m2y, dz = m1z - m2z;
  float similarity = sqrtf(dx * dx + dy * dy + dz * dz);
  float val = consistency + similarity;
#pragma unroll
  for (int off = 16; off > 0; off >>= 1) val += __shfl_xor(val, off, 32);
  if (bat == 0) out[0] = val;
}

extern "C" void kernel_launch(void* const* d_in, const int* in_sizes, int n_in,
                              void* d_out, int out_size, void* d_ws, size_t ws_size,
                              hipStream_t stream) {
  const float* x1 = (const float*)d_in[0];
  const float* x2 = (const float*)d_in[1];
  const int* b1 = (const int*)d_in[2];
  const int* b2 = (const int*)d_in[3];
  const int* f1 = (const int*)d_in[4];
  const int* f2 = (const int*)d_in[5];
  int F = in_sizes[4] / 3;
  float* ws = (float*)d_ws;
  float* out = (float*)d_out;

  hipLaunchKernelGGL(fn_init, dim3(4), dim3(256), 0, stream, ws);
  hipLaunchKernelGGL(fn_main, dim3(MAIN_BLOCKS), dim3(BLOCK_SIZE), 0, stream,
                     x1, x2, b1, b2, f1, f2, ws, F);
  hipLaunchKernelGGL(fn_final, dim3(1), dim3(32), 0, stream, ws, out);
}
